// MultiHeadAttention_68238440398882
// MI455X (gfx1250) — compile-verified, matrix operand order checked by reading, not run on hardware
//
#include <hip/hip_runtime.h>
#include <hip/hip_bf16.h>

#define D_MODEL 1024
#define N_HEADS 16
#define BATCH   4
#define SEQ     2048
#define HDIM    64   // D_MODEL / N_HEADS

#define AS1 __attribute__((address_space(1)))
#define AS3 __attribute__((address_space(3)))

#if defined(__HIP_DEVICE_COMPILE__) && __has_builtin(__builtin_amdgcn_global_load_async_to_lds_b128)
#define USE_ASYNC_LDS 1
#else
#define USE_ASYNC_LDS 0
#endif

typedef __attribute__((ext_vector_type(16))) __bf16 v16bf;
typedef __attribute__((ext_vector_type(8)))  __bf16 v8bf;
typedef __attribute__((ext_vector_type(8)))  float  v8f;
typedef __attribute__((ext_vector_type(4)))  float  v4f;
typedef __attribute__((ext_vector_type(4)))  int    v4i;
typedef __attribute__((ext_vector_type(2)))  unsigned long long v2u;

union Frag16 { v16bf v; v8bf h[2]; __bf16 e[16]; };
union Mask16 { v2u v; unsigned char c[16]; };
union Mask8  { unsigned long long u; unsigned char c[8]; };

static __device__ inline v16bf make_frag(v8bf lo, v8bf hi) {
  Frag16 f; f.h[0] = lo; f.h[1] = hi; return f.v;
}

#if USE_ASYNC_LDS
static __device__ inline void async_copy16(const void* g, void* l) {
  // signature (from compiler diagnostic): param0 = v4i in AS1 (__device__),
  // dst = LDS pointer, then imm offset / imm cpol.
  __builtin_amdgcn_global_load_async_to_lds_b128((AS1 v4i*)g, (AS3 v4i*)l, 0, 0);
}
static __device__ inline void wait_async() {
#if __has_builtin(__builtin_amdgcn_s_wait_asynccnt)
  __builtin_amdgcn_s_wait_asynccnt(0);
#else
  asm volatile("s_wait_asynccnt 0" ::: "memory");
#endif
}
#endif

// ---------------------------------------------------------------------------
// fp32 -> bf16 elementwise convert (weights only; activations are fused)
// ---------------------------------------------------------------------------
__global__ void cvt_f32_bf16(const float* __restrict__ in, __bf16* __restrict__ out, int n) {
  int i = blockIdx.x * blockDim.x + threadIdx.x;
  if (i < n) out[i] = (__bf16)in[i];
}

// ---------------------------------------------------------------------------
// 128x128-tile GEMM with double-buffered LDS: C = A[M,1024]*B[1024,N] + bias
// 256 threads = 8 wave32 waves, each wave computes 32x64 via WMMA bf16.
// ---------------------------------------------------------------------------
template<bool A_F32, bool OUT_F32>
__global__ __launch_bounds__(256)
void gemm128(const void* __restrict__ Ap, const __bf16* __restrict__ Bw,
             const float* __restrict__ bias, __bf16* __restrict__ Cb,
             float* __restrict__ Cf) {
  __shared__ __bf16 As[2][128 * 32];   // [m][k]  16 KB
  __shared__ __bf16 Bs[2][128 * 32];   // [n][k]  16 KB (transposed)

  const int tid  = threadIdx.x;
  const int lane = tid & 31;
  const int wave = tid >> 5;
  const int wm = wave >> 1;            // M offset wm*32
  const int wn = wave & 1;             // N offset wn*64
  const int ln = lane & 15;
  const int hi = lane >> 4;
  const int m0 = blockIdx.y * 128;
  const int n0 = blockIdx.x * 128;

  v8f acc[2][4] = {};

  const int arow = tid >> 1, akc = (tid & 1) * 16;   // A stage mapping
  const int bkr  = tid >> 3, bnc = (tid & 7) * 16;   // B stage mapping

  auto stageA = [&](int buf, int kt) {
    const int k0 = kt * 32;
    if (A_F32) {
      const float* src = (const float*)Ap + (size_t)(m0 + arow) * D_MODEL + k0 + akc;
      v4f f0 = *(const v4f*)(src + 0);
      v4f f1 = *(const v4f*)(src + 4);
      v4f f2 = *(const v4f*)(src + 8);
      v4f f3 = *(const v4f*)(src + 12);
      __bf16* dst = &As[buf][arow * 32 + akc];
#pragma unroll
      for (int j = 0; j < 4; ++j) {
        dst[0  + j] = (__bf16)f0[j];
        dst[4  + j] = (__bf16)f1[j];
        dst[8  + j] = (__bf16)f2[j];
        dst[12 + j] = (__bf16)f3[j];
      }
    } else {
      const __bf16* src = (const __bf16*)Ap + (size_t)(m0 + arow) * D_MODEL + k0 + akc;
      __bf16* dst = &As[buf][arow * 32 + akc];
#if USE_ASYNC_LDS
      async_copy16(src, dst);
      async_copy16(src + 8, dst + 8);
#else
      *(v8bf*)dst       = *(const v8bf*)src;
      *(v8bf*)(dst + 8) = *(const v8bf*)(src + 8);
#endif
    }
  };
  auto stageB = [&](int buf, int kt) {
    const __bf16* src = Bw + (size_t)(kt * 32 + bkr) * D_MODEL + n0 + bnc;
    Frag16 t; t.h[0] = *(const v8bf*)src; t.h[1] = *(const v8bf*)(src + 8);
#pragma unroll
    for (int j = 0; j < 16; ++j) Bs[buf][(bnc + j) * 32 + bkr] = t.e[j];
  };

  stageA(0, 0);
  stageB(0, 0);
#if USE_ASYNC_LDS
  if (!A_F32) wait_async();
#endif
  __syncthreads();

  for (int kt = 0; kt < D_MODEL / 32; ++kt) {
    const int cur = kt & 1, nxt = cur ^ 1;
    if (kt + 1 < D_MODEL / 32) { stageA(nxt, kt + 1); stageB(nxt, kt + 1); }
    if (kt + 2 < D_MODEL / 32) {
      const int kp = (kt + 2) * 32;
      if (A_F32)
        __builtin_prefetch((const float*)Ap + (size_t)(m0 + arow) * D_MODEL + kp + akc, 0, 1);
      else
        __builtin_prefetch((const __bf16*)Ap + (size_t)(m0 + arow) * D_MODEL + kp + akc, 0, 1);
      __builtin_prefetch(Bw + (size_t)(kp + bkr) * D_MODEL + n0 + bnc, 0, 1);
    }

    v16bf af[2], bfm[4];
#pragma unroll
    for (int i = 0; i < 2; ++i) {
      const __bf16* p = &As[cur][(wm * 32 + i * 16 + ln) * 32 + 8 * hi];
      af[i] = make_frag(*(const v8bf*)p, *(const v8bf*)(p + 16));
    }
#pragma unroll
    for (int j = 0; j < 4; ++j) {
      const __bf16* p = &Bs[cur][(wn * 64 + j * 16 + ln) * 32 + 8 * hi];
      bfm[j] = make_frag(*(const v8bf*)p, *(const v8bf*)(p + 16));
    }
#pragma unroll
    for (int i = 0; i < 2; ++i)
#pragma unroll
      for (int j = 0; j < 4; ++j)
        acc[i][j] = __builtin_amdgcn_wmma_f32_16x16x32_bf16(
            false, af[i], false, bfm[j], (short)0, acc[i][j], false, false);

#if USE_ASYNC_LDS
    if (!A_F32) wait_async();
#endif
    __syncthreads();
  }

  // epilogue: +bias, store
#pragma unroll
  for (int i = 0; i < 2; ++i)
#pragma unroll
    for (int j = 0; j < 4; ++j) {
      const int col = n0 + wn * 64 + j * 16 + ln;
      const float bv = bias[col];
#pragma unroll
      for (int r = 0; r < 8; ++r) {
        const int row = m0 + wm * 32 + i * 16 + r + 8 * hi;
        const float val = acc[i][j][r] + bv;
        const size_t idx = (size_t)row * D_MODEL + col;
        if (OUT_F32) Cf[idx] = val; else Cb[idx] = (__bf16)val;
      }
    }
}

// ---------------------------------------------------------------------------
// Flash attention: workgroup = (batch, head, 64 q-rows), 4 waves, 32-key
// blocks, double-buffered K (async-to-LDS), V (transposed), mask (transposed).
// ---------------------------------------------------------------------------
__global__ __launch_bounds__(128)
void attention(const __bf16* __restrict__ Qp, const __bf16* __restrict__ Kp,
               const __bf16* __restrict__ Vp, const unsigned char* __restrict__ mask,
               __bf16* __restrict__ Op) {
#if USE_ASYNC_LDS
  __shared__ __bf16 Ks[2][32 * HDIM];       // row-major [key][d], 8 KB
#endif
  __shared__ __bf16 Vt[2][HDIM * 32];       // transposed [d][key], 8 KB
  __shared__ unsigned char Mt[2][32 * 64];  // transposed [col][qrow], 4 KB
  __shared__ __bf16 Pb[4][16 * 32];         // wave-private P, 4 KB

  const int tid  = threadIdx.x;
  const int lane = tid & 31;
  const int wave = tid >> 5;
  const int ln = lane & 15;
  const int hi = lane >> 4;
  const int b = blockIdx.z, h = blockIdx.y;
  const int qb = blockIdx.x * 64;
  const int q0 = qb + wave * 16;
  const size_t mrow0 = (((size_t)b * N_HEADS + h) * SEQ + qb) * SEQ;

  // Q fragments kept in VGPRs: 16 rows x 64 dims = 2 A-fragments
  v16bf aq[2];
#pragma unroll
  for (int c = 0; c < 2; ++c) {
    const __bf16* p = Qp + ((size_t)(b * SEQ + q0 + ln)) * D_MODEL + h * HDIM + c * 32 + 8 * hi;
    aq[c] = make_frag(*(const v8bf*)p, *(const v8bf*)(p + 16));
  }

  v8f o[4] = {};
  float mi[8], li[8];
#pragma unroll
  for (int r = 0; r < 8; ++r) { mi[r] = -3.0e38f; li[r] = 0.0f; }

  // staging thread mappings
  const int vkey = tid >> 2, vd0 = (tid & 3) * 16;   // V transpose / K async
  const int mrow = tid >> 1, mc0 = (tid & 1) * 16;   // mask transpose

  auto kvbase = [&](int kb, int row) {
    return ((size_t)(b * SEQ + kb * 32 + row)) * D_MODEL + h * HDIM;
  };

#if USE_ASYNC_LDS
  auto stageK = [&](int buf, int kb) {   // async global -> LDS, layout preserved
    const __bf16* src = Kp + kvbase(kb, vkey) + vd0;
    __bf16* dst = &Ks[buf][vkey * HDIM + vd0];
    async_copy16(src, dst);
    async_copy16(src + 8, dst + 8);
  };
#endif
  auto stageV = [&](int buf, int kb) {   // transpose through registers
    const __bf16* p = Vp + kvbase(kb, vkey) + vd0;
    Frag16 t; t.h[0] = *(const v8bf*)p; t.h[1] = *(const v8bf*)(p + 8);
#pragma unroll
    for (int j = 0; j < 16; ++j) Vt[buf][(vd0 + j) * 32 + vkey] = t.e[j];
  };
  auto stageM = [&](int buf, int kb) {   // transpose 64x32 mask tile
    const unsigned char* ms = mask + mrow0 + (size_t)mrow * SEQ + kb * 32 + mc0;
    Mask16 t; t.v = *(const v2u*)ms;
#pragma unroll
    for (int j = 0; j < 16; ++j) Mt[buf][(mc0 + j) * 64 + mrow] = t.c[j];
  };

#if USE_ASYNC_LDS
  stageK(0, 0);
#endif
  stageV(0, 0);
  stageM(0, 0);
#if USE_ASYNC_LDS
  wait_async();
#endif
  __syncthreads();

  for (int kb = 0; kb < SEQ / 32; ++kb) {
    const int cur = kb & 1, nxt = cur ^ 1;
    if (kb + 1 < SEQ / 32) {
#if USE_ASYNC_LDS
      stageK(nxt, kb + 1);
#endif
      stageV(nxt, kb + 1);
      stageM(nxt, kb + 1);
    }
    if (kb + 2 < SEQ / 32) {
      __builtin_prefetch(Kp + kvbase(kb + 2, vkey) + vd0, 0, 1);
      __builtin_prefetch(Vp + kvbase(kb + 2, vkey) + vd0, 0, 1);
      __builtin_prefetch(mask + mrow0 + (size_t)mrow * SEQ + (kb + 2) * 32 + mc0, 0, 1);
    }

    // ---- S = Q * K^T
    v8f s[2] = {};
#pragma unroll
    for (int nt = 0; nt < 2; ++nt)
#pragma unroll
      for (int c = 0; c < 2; ++c) {
#if USE_ASYNC_LDS
        const __bf16* p = &Ks[cur][(nt * 16 + ln) * HDIM + c * 32 + 8 * hi];
#else
        const __bf16* p = Kp + kvbase(kb, nt * 16 + ln) + c * 32 + 8 * hi;
#endif
        v16bf bk = make_frag(*(const v8bf*)p, *(const v8bf*)(p + 16));
        s[nt] = __builtin_amdgcn_wmma_f32_16x16x32_bf16(
            false, aq[c], false, bk, (short)0, s[nt], false, false);
      }

    // ---- scale (1/sqrt(64)) and mask (one ds_load_b64 per fragment)
#pragma unroll
    for (int nt = 0; nt < 2; ++nt) {
      Mask8 mk;
      mk.u = *(const unsigned long long*)&Mt[cur][(nt * 16 + ln) * 64 + wave * 16 + 8 * hi];
#pragma unroll
      for (int r = 0; r < 8; ++r) {
        float v = s[nt][r] * 0.125f;
        if (mk.c[r]) v = -1.0e10f;
        s[nt][r] = v;
      }
    }

    // ---- online softmax (row spans 16 lanes of one half-wave)
    float alpha[8];
#pragma unroll
    for (int r = 0; r < 8; ++r) {
      float rm = fmaxf(s[0][r], s[1][r]);
#pragma unroll
      for (int ofs = 1; ofs < 16; ofs <<= 1)
        rm = fmaxf(rm, __shfl_xor(rm, ofs, 16));
      const float mn = fmaxf(mi[r], rm);
      alpha[r] = __expf(mi[r] - mn);
      s[0][r] = __expf(s[0][r] - mn);
      s[1][r] = __expf(s[1][r] - mn);
      float rs = s[0][r] + s[1][r];
#pragma unroll
      for (int ofs = 1; ofs < 16; ofs <<= 1)
        rs += __shfl_xor(rs, ofs, 16);
      li[r] = li[r] * alpha[r] + rs;
      mi[r] = mn;
    }
#pragma unroll
    for (int dt = 0; dt < 4; ++dt)
#pragma unroll
      for (int r = 0; r < 8; ++r)
        o[dt][r] *= alpha[r];

    // ---- P (16x32) -> wave-private LDS, reshape to A-fragment; O += P*V
#pragma unroll
    for (int nt = 0; nt < 2; ++nt)
#pragma unroll
      for (int r = 0; r < 8; ++r)
        Pb[wave][(r + 8 * hi) * 32 + nt * 16 + ln] = (__bf16)s[nt][r];
    {
      const __bf16* p = &Pb[wave][ln * 32 + 8 * hi];
      v16bf ap = make_frag(*(const v8bf*)p, *(const v8bf*)(p + 16));
#pragma unroll
      for (int dt = 0; dt < 4; ++dt) {
        const __bf16* q = &Vt[cur][(dt * 16 + ln) * 32 + 8 * hi];
        v16bf bv = make_frag(*(const v8bf*)q, *(const v8bf*)(q + 16));
        o[dt] = __builtin_amdgcn_wmma_f32_16x16x32_bf16(
            false, ap, false, bv, (short)0, o[dt], false, false);
      }
    }

#if USE_ASYNC_LDS
    wait_async();
#endif
    __syncthreads();
  }

  // ---- normalize and store merged-head layout [b, s, h*64+d] as bf16
#pragma unroll
  for (int r = 0; r < 8; ++r) {
    const float inv = __builtin_amdgcn_rcpf(li[r]);
    const int row = q0 + r + 8 * hi;
#pragma unroll
    for (int dt = 0; dt < 4; ++dt) {
      const float val = o[dt][r] * inv;
      Op[((size_t)(b * SEQ + row)) * D_MODEL + h * HDIM + dt * 16 + ln] = (__bf16)val;
    }
  }
}

// ---------------------------------------------------------------------------
extern "C" void kernel_launch(void* const* d_in, const int* in_sizes, int n_in,
                              void* d_out, int out_size, void* d_ws, size_t ws_size,
                              hipStream_t stream) {
  (void)in_sizes; (void)n_in; (void)out_size; (void)ws_size;
  const float* q   = (const float*)d_in[0];
  const float* k   = (const float*)d_in[1];
  const float* v   = (const float*)d_in[2];
  const unsigned char* mask = (const unsigned char*)d_in[3];
  const float* W_q = (const float*)d_in[4];
  const float* b_q = (const float*)d_in[5];
  const float* W_o = (const float*)d_in[6];
  const float* b_o = (const float*)d_in[7];
  float* out = (float*)d_out;

  // workspace: [0,2M) W_q bf16 | [2M,4M) W_o bf16 | [4M,+48M) QKV proj bf16
  //            | next 16M: attention output bf16
  char* ws = (char*)d_ws;
  const size_t ACT = (size_t)BATCH * SEQ * D_MODEL;   // 8388608 elements
  __bf16* Wq_bf = (__bf16*)(ws);
  __bf16* Wo_bf = (__bf16*)(ws + (size_t)2 * 1024 * 1024);
  __bf16* qkvp  = (__bf16*)(ws + (size_t)4 * 1024 * 1024);
  __bf16* attn  = (__bf16*)(ws + (size_t)4 * 1024 * 1024 + 3 * ACT * 2);

  const int nW = D_MODEL * D_MODEL;
  cvt_f32_bf16<<<dim3((nW + 255) / 256), dim3(256), 0, stream>>>(W_q, Wq_bf, nW);
  cvt_f32_bf16<<<dim3((nW + 255) / 256), dim3(256), 0, stream>>>(W_o, Wo_bf, nW);

  const dim3 gg(D_MODEL / 128, (BATCH * SEQ) / 128);  // (8, 64)
  const float* xin[3] = {q, k, v};
  for (int t = 0; t < 3; ++t) {
    gemm128<true, false><<<gg, dim3(256), 0, stream>>>(
        (const void*)xin[t], Wq_bf, b_q, qkvp + (size_t)t * ACT, nullptr);
  }

  attention<<<dim3(SEQ / 64, N_HEADS, BATCH), dim3(128), 0, stream>>>(
      qkvp, qkvp + ACT, qkvp + 2 * ACT, mask, attn);

  gemm128<false, true><<<gg, dim3(256), 0, stream>>>(
      (const void*)attn, Wo_bf, b_o, nullptr, out);
}
